// LSTM_24936580121326
// MI455X (gfx1250) — compile-verified, matrix operand order checked by reading, not run on hardware
//
#include <hip/hip_runtime.h>
#include <hip/hip_bf16.h>

typedef __attribute__((ext_vector_type(16))) _Float16 v16h;
typedef __attribute__((ext_vector_type(2)))  __fp16   v2fp16;   // cvt_pkrtz result type
typedef __attribute__((ext_vector_type(8)))  float    v8f;
typedef __attribute__((ext_vector_type(8)))  int      v8i;
typedef __attribute__((ext_vector_type(2)))  unsigned v2u;

#define T_LEN 49
#define HID   8
#define U     2          // batch tiles per wave (independent chains for ILP)

// ---- gfx1250 hardware tanh (1 TRANS op), else exp2+rcp emulation ----
#if __has_builtin(__builtin_amdgcn_tanhf)
  #define HAVE_HW_TANH 1
  __device__ __forceinline__ float fast_tanh(float x) { return __builtin_amdgcn_tanhf(x); }
#elif __has_builtin(__builtin_amdgcn_tanh_f32)
  #define HAVE_HW_TANH 1
  __device__ __forceinline__ float fast_tanh(float x) { return __builtin_amdgcn_tanh_f32(x); }
#else
  #define HAVE_HW_TANH 0
#endif

// ---- VALU half-swap (v_permlane16_swap_b32) if available, else DS shuffle ----
#if __has_builtin(__builtin_amdgcn_permlane16_swap)
  #define HAVE_SWAP 1
  // swap(a,b): a_new = [a_lo | b_lo], b_new = [a_hi | b_hi]
  __device__ __forceinline__ void half_swap_u(unsigned& a, unsigned& b) {
      v2u r = __builtin_bit_cast(v2u,
                __builtin_amdgcn_permlane16_swap(a, b, false, false));
      a = r[0]; b = r[1];
  }
  __device__ __forceinline__ void half_swap_f(float& a, float& b) {
      unsigned au = __builtin_bit_cast(unsigned, a);
      unsigned bu = __builtin_bit_cast(unsigned, b);
      half_swap_u(au, bu);
      a = __builtin_bit_cast(float, au);
      b = __builtin_bit_cast(float, bu);
  }
#else
  #define HAVE_SWAP 0
#endif

__device__ __forceinline__ float sigm_exp(float x) {
    return __builtin_amdgcn_rcpf(1.0f + __builtin_amdgcn_exp2f(-1.44269504088896f * x));
}

#if HAVE_HW_TANH
__device__ __forceinline__ float sigm(float x) {      // 0.5*tanh(x/2)+0.5, 1 TRANS op
    return fmaf(0.5f, fast_tanh(0.5f * x), 0.5f);
}
#else
__device__ __forceinline__ float sigm(float x) { return sigm_exp(x); }
#endif

__device__ __forceinline__ v8f wmma_f16(v16h a, const v8i& b, v8f c) {
    // D = A(16x32 f16) * B(32x16 f16) + C(16x16 f32)
    return __builtin_amdgcn_wmma_f32_16x16x32_f16(false, a, false,
                                                  __builtin_bit_cast(v16h, b),
                                                  (short)0, c, false, false);
}

// d0: gates 0..15  -> reg r: i_r (lanes 0-15), f_r (lanes 16-31)   [both sigmoid]
// d1: gates 16..31 -> reg r: g_r (lanes 0-15, tanh), o_r (lanes 16-31, sigmoid)
// sB/sA/sC: per-lane consts so a1 = sA*tanh(sB*x)+sC is tanh (lo) or sigmoid (hi).
// c[] consumed at hi lanes only (replicated when HAVE_SWAP).
// Updates the NEXT-STEP B matrix IN PLACE: h pairs packed to f16 at hi lanes,
// moved to lo lanes (one 32-bit half-swap/shuffle per pair), hi lanes zeroed.
// Elements 4..7 of *Bout (K=8..15 pad) are never touched -> stay zero.
__device__ __forceinline__ void cell_update(const v8f& d0, const v8f& d1,
                                            bool hi, float sB, float sA, float sC,
                                            float* c, v8i* Bout) {
#pragma unroll
    for (int v = 0; v < 4; ++v) {
        float a1p[2], prod[2];
#pragma unroll
        for (int k = 0; k < 2; ++k) {
            const int r = 2 * v + k;
            float a0 = sigm(d0[r]);                        // i (lo) / f (hi)
#if HAVE_HW_TANH
            float a1 = fmaf(sA, fast_tanh(sB * d1[r]), sC);   // g (lo) / o (hi)
#else
            float a1 = fmaf(sA, sigm_exp(sB * d1[r]), sC);
#endif
            a1p[k]  = a1;
            prod[k] = a0 * (hi ? c[r] : a1);               // lo: i*g   hi: f*c
        }
        float cn0, cn1;
#if HAVE_SWAP
        float x = prod[0], y = prod[1];
        half_swap_f(x, y);            // x=[p0_lo|p1_lo]  y=[p0_hi|p1_hi]
        float z = x + y;              // [cn0 @lo | cn1 @hi]
        cn0 = z; cn1 = z;
        half_swap_f(cn0, cn1);        // cn0, cn1 each replicated across halves
#else
        cn0 = prod[0] + __shfl_xor(prod[0], 16);
        cn1 = prod[1] + __shfl_xor(prod[1], 16);
#endif
        c[2 * v] = cn0; c[2 * v + 1] = cn1;
#if HAVE_HW_TANH
        float th0 = fast_tanh(cn0), th1 = fast_tanh(cn1);
#else
        float th0 = fmaf(2.0f, sigm_exp(2.0f * cn0), -1.0f);
        float th1 = fmaf(2.0f, sigm_exp(2.0f * cn1), -1.0f);
#endif
        float hx0 = th0 * a1p[0];                          // h valid at hi lanes
        float hx1 = th1 * a1p[1];
        v2fp16 p = __builtin_amdgcn_cvt_pkrtz(hx0, hx1);   // pack pair -> f16x2 (hi lanes)
        unsigned pi = __builtin_bit_cast(unsigned, p);
#if HAVE_SWAP
        unsigned pa = pi, pb = pi;
        half_swap_u(pa, pb);          // pb = [pi_hi | pi_hi] : packed h replicated
        (*Bout)[v] = hi ? 0 : (int)pb;
#else
        int ps = __shfl_xor((int)pi, 16);
        (*Bout)[v] = hi ? 0 : ps;
#endif
    }
}

__global__ __launch_bounds__(256)
void lstm_fused_kernel(const float* __restrict__ X,
                       const float* __restrict__ Wih0, const float* __restrict__ Whh0,
                       const float* __restrict__ bi0,  const float* __restrict__ bh0,
                       const float* __restrict__ Wih1, const float* __restrict__ Whh1,
                       const float* __restrict__ bi1,  const float* __restrict__ bh1,
                       const float* __restrict__ Wlin, const float* __restrict__ blin,
                       float* __restrict__ out, int Bn) {
    __shared__ float sX[8][U * 784 + 16];   // 2 tiles x 16 x 49 floats per wave

    const int  tid  = threadIdx.x;
    const int  wave = tid >> 5;
    const int  lane = tid & 31;
    const int  col  = lane & 15;            // batch column within tile
    const bool hi   = lane >= 16;
    const long pair = (long)blockIdx.x * 8 + wave;
    const long b0   = pair * (16 * U);

    // ---- stage this wave's 2 x (16 x T) input block into LDS (coalesced) ----
    if (b0 + 16 * U - 1 < Bn) {
        const float* src = X + (size_t)b0 * T_LEN;
        for (int i = lane; i < 16 * U * T_LEN; i += 32) sX[wave][i] = src[i];
    }
    __syncthreads();
    if (b0 + 16 * U - 1 >= Bn) return;

    // ---- A-matrix weights (16x32 f16 layout, K=0..7 live; lanes>=16 & elems 8..15 zero) ----
    const float amask = hi ? 0.0f : 1.0f;
    auto loadA = [&](const float* W, int rowbase) -> v16h {
        v16h a = {};
        const float* wr = W + (size_t)(rowbase + col) * HID;
#pragma unroll
        for (int e = 0; e < 8; ++e) a[e] = (_Float16)(wr[e] * amask);
        return a;
    };
    const v16h Ahh0_lo = loadA(Whh0, 0), Ahh0_hi = loadA(Whh0, 16);
    const v16h Ahh1_lo = loadA(Whh1, 0), Ahh1_hi = loadA(Whh1, 16);
    const v16h Aih1_lo = loadA(Wih1, 0), Aih1_hi = loadA(Wih1, 16);

    // ---- per-lane gate constants (gate index M = r + 8*hi within each block) ----
    const int goff = hi ? 8 : 0;
    float w0lo[8], bb0lo[8], w0hi[8], bb0hi[8];
    v8f CB0, CB1;                                  // layer-1 bias C matrices
#pragma unroll
    for (int r = 0; r < 8; ++r) {
        int gl = r + goff, gh = 16 + r + goff;
        w0lo[r]  = Wih0[gl];            w0hi[r]  = Wih0[gh];
        bb0lo[r] = bi0[gl] + bh0[gl];   bb0hi[r] = bi0[gh] + bh0[gh];
        CB0[r]   = bi1[gl] + bh1[gl];   CB1[r]   = bi1[gh] + bh1[gh];
    }
#if HAVE_HW_TANH
    const float sB = hi ? 0.5f : 1.0f;             // a1 = sA*tanh(sB*x)+sC
    const float sA = hi ? 0.5f : 1.0f;
    const float sC = hi ? 0.5f : 0.0f;
#else
    const float sB = hi ? 1.0f : 2.0f;             // a1 = sA*sigm(sB*x)+sC
    const float sA = hi ? 1.0f : 2.0f;
    const float sC = hi ? 0.0f : -1.0f;
#endif

    // ---- state (two independent tiles); B kept as v8i, upper half stays zero ----
    v8i B1[U], B2[U];
    float c1[U][8], c2[U][8];
#pragma unroll
    for (int u = 0; u < U; ++u) {
        B1[u] = (v8i){}; B2[u] = (v8i){};
#pragma unroll
        for (int r = 0; r < 8; ++r) { c1[u][r] = 0.0f; c2[u][r] = 0.0f; }
    }

    for (int t = 0; t < T_LEN; ++t) {
        // ---------- layer 0 ----------
        v8f C0[U], C1[U], D0[U], D1[U];
#pragma unroll
        for (int u = 0; u < U; ++u) {
            float x = sX[wave][u * 784 + col * T_LEN + t];
#pragma unroll
            for (int r = 0; r < 8; ++r) {
                C0[u][r] = fmaf(x, w0lo[r], bb0lo[r]);
                C1[u][r] = fmaf(x, w0hi[r], bb0hi[r]);
            }
        }
#pragma unroll
        for (int u = 0; u < U; ++u) {
            D0[u] = wmma_f16(Ahh0_lo, B1[u], C0[u]);   // gates 0..15
            D1[u] = wmma_f16(Ahh0_hi, B1[u], C1[u]);   // gates 16..31
        }
#pragma unroll
        for (int u = 0; u < U; ++u)
            cell_update(D0[u], D1[u], hi, sB, sA, sC, c1[u], &B1[u]);

        // ---------- layer 1 ----------
        v8f E0[U], E1[U];
#pragma unroll
        for (int u = 0; u < U; ++u) {
            E0[u] = wmma_f16(Aih1_lo, B1[u], CB0);     // bias + Wih1 * h1
            E1[u] = wmma_f16(Aih1_hi, B1[u], CB1);
        }
#pragma unroll
        for (int u = 0; u < U; ++u) {
            E0[u] = wmma_f16(Ahh1_lo, B2[u], E0[u]);   // + Whh1 * h2
            E1[u] = wmma_f16(Ahh1_hi, B2[u], E1[u]);
        }
#pragma unroll
        for (int u = 0; u < U; ++u)
            cell_update(E0[u], E1[u], hi, sB, sA, sC, c2[u], &B2[u]);
    }

    // ---- head: relu -> linear -> relu on last h2 (f16 pairs in B2, valid at lo lanes) ----
#pragma unroll
    for (int u = 0; u < U; ++u) {
        float acc = blin[0];
#pragma unroll
        for (int v = 0; v < 4; ++v) {
            v2fp16 p = __builtin_bit_cast(v2fp16, B2[u][v]);
            acc = fmaf(fmaxf((float)p[0], 0.0f), Wlin[2 * v],     acc);
            acc = fmaf(fmaxf((float)p[1], 0.0f), Wlin[2 * v + 1], acc);
        }
        float y = fmaxf(acc, 0.0f);
        if (!hi) out[b0 + u * 16 + col] = y;
    }
}

extern "C" void kernel_launch(void* const* d_in, const int* in_sizes, int n_in,
                              void* d_out, int out_size, void* d_ws, size_t ws_size,
                              hipStream_t stream) {
    (void)n_in; (void)out_size; (void)d_ws; (void)ws_size;
    const float* X    = (const float*)d_in[0];
    const float* Wih0 = (const float*)d_in[1];
    const float* Whh0 = (const float*)d_in[2];
    const float* bi0  = (const float*)d_in[3];
    const float* bh0  = (const float*)d_in[4];
    const float* Wih1 = (const float*)d_in[5];
    const float* Whh1 = (const float*)d_in[6];
    const float* bi1  = (const float*)d_in[7];
    const float* bh1  = (const float*)d_in[8];
    const float* Wlin = (const float*)d_in[9];
    const float* blin = (const float*)d_in[10];
    float* out = (float*)d_out;

    const int Bn     = in_sizes[0] / T_LEN;           // 131072
    const int pairs  = (Bn + 16 * U - 1) / (16 * U);  // 4096 waves
    const int blocks = (pairs + 7) / 8;               // 8 waves (256 threads) per block

    lstm_fused_kernel<<<blocks, 256, 0, stream>>>(X, Wih0, Whh0, bi0, bh0,
                                                  Wih1, Whh1, bi1, bh1,
                                                  Wlin, blin, out, Bn);
}
